// LSTM_48902497632321
// MI455X (gfx1250) — matrix operand order repaired and checked by reading, not compiled, not run on hardware
//
#include <hip/hip_runtime.h>

// ---------------------------------------------------------------------------
// 2-layer LSTM (B=512, S=512, D=64, H=128) + final linear, fused scan.
// Each block owns 16 batch rows and runs the full 512-step, 2-layer sequence.
// Gates GEMMs use v_wmma_f32_16x16x32_bf16; cell math is per-lane VALU.
// Weight B-tiles are software-pipelined one k-group ahead (cur/nxt double
// buffer); __builtin_amdgcn_sched_barrier pins the prefetch loads above the
// consuming WMMAs so the scheduler cannot re-serialize load->wait(0)->wmma.
// ---------------------------------------------------------------------------

typedef __attribute__((ext_vector_type(16))) __bf16 v16bf;
typedef __attribute__((ext_vector_type(8)))  float  v8f;

#define WMMA_BF16(a, b, c) \
  __builtin_amdgcn_wmma_f32_16x16x32_bf16(false, (a), false, (b), (short)0, (c), false, false)

#if defined(__has_builtin) && __has_builtin(__builtin_amdgcn_sched_barrier)
#define SCHED_FENCE() __builtin_amdgcn_sched_barrier(0)
#else
#define SCHED_FENCE()
#endif

union V16U { uint4 u[2]; v16bf v; };

__device__ __forceinline__ unsigned short f2bf(float f) {
  unsigned u = __float_as_uint(f);
  unsigned r = (u + 0x7FFFu + ((u >> 16) & 1u)) >> 16;   // round-to-nearest-even
  return (unsigned short)r;
}
__device__ __forceinline__ float bf2f(unsigned short s) {
  return __uint_as_float(((unsigned)s) << 16);
}

__device__ __forceinline__ float sigmoidf_(float x) {
  return 1.0f / (1.0f + __expf(-x));
}
__device__ __forceinline__ float tanhf_(float x) {
  float e = __expf(2.0f * x);
  return 1.0f - 2.0f / (e + 1.0f);
}

// Opaque zero: compiler cannot prove it is 0 or loop-invariant, so loads whose
// address includes it cannot be hoisted out of the time-step loop. Kept in an
// SGPR so weight loads remain global_load_b128 with scalar-derived bases.
__device__ __forceinline__ size_t opaque_zero() {
  size_t z = 0;
  asm volatile("" : "+s"(z));
  return z;
}

// B-matrix (KxN = 32x16) tile of W^T, i.e. B[k][n] = W[n][k], W row-major (ldk).
// k order used here (same as A): lanes 0-15 hold K=0..15 (2/VGPR), lanes 16-31 hold K=16..31.
// => lane reads 16 consecutive bf16 from row n starting at kt + 16*(lane>=16).
__device__ __forceinline__ v16bf load_b_global(const unsigned short* __restrict__ W,
                                               size_t z, int ldk, int nbase,
                                               int kt, int lane) {
  int n  = nbase + (lane & 15);
  int k0 = kt + ((lane >> 4) << 4);
  const unsigned short* p = W + z + (size_t)n * ldk + k0;
  V16U t;
  t.u[0] = *(const uint4*)(p);
  t.u[1] = *(const uint4*)(p + 8);
  return t.v;
}

// A-matrix (MxK = 16x32) tile from row-major (ld, K contiguous) bf16 buffer.
// k order used here (same as B): lanes 0-15: V0-7 = K kt..kt+15;
//                lanes 16-31: V0-7 = K kt+16..kt+31.
__device__ __forceinline__ v16bf load_a_lds(const unsigned short* base,
                                            int ld, int kt, int lane) {
  int m  = lane & 15;
  int hi = lane >> 4;
  const unsigned short* p = base + m * ld + kt + 16 * hi;
  V16U t;
  t.u[0] = *(const uint4*)(p);
  t.u[1] = *(const uint4*)(p + 8);
  return t.v;
}

// Four gate B-tiles (i,f,g,o) for one 32-wide k-group.
struct B4 { v16bf b0, b1, b2, b3; };

__device__ __forceinline__ B4 loadB4(const unsigned short* __restrict__ W,
                                     size_t z, int ldk, int wcol, int kt, int lane) {
  B4 r;
  r.b0 = load_b_global(W, z, ldk, 0 * 128 + wcol, kt, lane);
  r.b1 = load_b_global(W, z, ldk, 1 * 128 + wcol, kt, lane);
  r.b2 = load_b_global(W, z, ldk, 2 * 128 + wcol, kt, lane);
  r.b3 = load_b_global(W, z, ldk, 3 * 128 + wcol, kt, lane);
  return r;
}

__device__ __forceinline__ void wmma4(v8f acc[4], v16bf a, const B4& B) {
  acc[0] = WMMA_BF16(a, B.b0, acc[0]);
  acc[1] = WMMA_BF16(a, B.b1, acc[1]);
  acc[2] = WMMA_BF16(a, B.b2, acc[2]);
  acc[3] = WMMA_BF16(a, B.b3, acc[3]);
}

// LSTM cell nonlinearity on one wave's 16x16 slice (8 elems / lane).
__device__ __forceinline__ void cell_update(v8f acc[4], v8f& c, v8f& h) {
#pragma unroll
  for (int e = 0; e < 8; ++e) {
    float iv = sigmoidf_(acc[0][e]);
    float fv = sigmoidf_(acc[1][e]);
    float gv = tanhf_(acc[2][e]);
    float ov = sigmoidf_(acc[3][e]);
    float cv = fv * c[e] + iv * gv;
    c[e] = cv;
    h[e] = ov * tanhf_(cv);
  }
}

// ---------------------------------------------------------------------------
// Weight f32 -> bf16 conversion into workspace (one-time, fully parallel).
// Regions: W_ih0 (512x64), W_hh0 (512x128), W_ih1 (512x128), W_hh1 (512x128).
// ---------------------------------------------------------------------------
__global__ __launch_bounds__(256) void cvt_weights(const float* __restrict__ w0,
                                                   const float* __restrict__ w1,
                                                   const float* __restrict__ w2,
                                                   const float* __restrict__ w3,
                                                   unsigned short* __restrict__ out) {
  int i = blockIdx.x * 256 + threadIdx.x;
  float v;
  if (i < 32768)        v = w0[i];
  else if (i < 98304)   v = w1[i - 32768];
  else if (i < 163840)  v = w2[i - 98304];
  else                  v = w3[i - 163840];
  out[i] = f2bf(v);
}

// ---------------------------------------------------------------------------
// Fused 2-layer LSTM scan + final linear.
// Block: 256 threads = 8 wave32s. Wave w owns h-columns [w*16, w*16+16).
// Gate tiles for wave w sit at gate columns q*128 + w*16, q in {i,f,g,o}.
// C/D f32 layout: element (m,n): vgpr r = m%8, lane = n + 16*(m/8).
// ---------------------------------------------------------------------------
__global__ __launch_bounds__(256) void lstm_scan(
    const float* __restrict__ x,
    const unsigned short* __restrict__ Wih0, const unsigned short* __restrict__ Whh0,
    const unsigned short* __restrict__ Wih1, const unsigned short* __restrict__ Whh1,
    const float* __restrict__ bih0, const float* __restrict__ bhh0,
    const float* __restrict__ bih1, const float* __restrict__ bhh1,
    const float* __restrict__ Wlin, const float* __restrict__ blin,
    float* __restrict__ out) {
  __shared__ unsigned short xbuf[16 * 64];     // x_t tile, bf16 row-major
  __shared__ unsigned short h0buf[16 * 128];   // layer-0 hidden state, bf16
  __shared__ unsigned short h1buf[16 * 128];   // layer-1 hidden state, bf16

  const int tid    = threadIdx.x;
  const int lane   = tid & 31;
  const int w      = tid >> 5;        // wave id 0..7
  const int wcol   = w * 16;          // this wave's h-column base
  const int batch0 = blockIdx.x * 16; // 32 blocks cover B=512

  // zero initial hidden state
  for (int i = tid; i < 16 * 128; i += 256) { h0buf[i] = 0; h1buf[i] = 0; }

  const int ncol = wcol + (lane & 15);       // this lane's h-column (0..127)
  float bias0[4], bias1[4];
#pragma unroll
  for (int q = 0; q < 4; ++q) {
    int n = q * 128 + ncol;
    bias0[q] = bih0[n] + bhh0[n];
    bias1[q] = bih1[n] + bhh1[n];
  }

  v8f c0; v8f c1;
#pragma unroll
  for (int e = 0; e < 8; ++e) { c0[e] = 0.f; c1[e] = 0.f; }

  __syncthreads();

  // Prime the B-tile pipeline: first k-group of layer 0 (W_ih0, kt=0).
  B4 nxt = loadB4(Wih0, opaque_zero(), 64, wcol, 0, lane);

  // One pipelined k-group: prefetch group i+1, fence, then consume group i.
#define PIPE_STEP(Wn, ldkn, ktn, Abuf, Ald, Akt)                  \
  do {                                                            \
    B4 cur = nxt;                                                 \
    nxt = loadB4(Wn, z, ldkn, wcol, ktn, lane);                   \
    v16bf a_ = load_a_lds(Abuf, Ald, Akt, lane);                  \
    SCHED_FENCE(); /* keep prefetch loads above these WMMAs */    \
    wmma4(acc, a_, cur);                                          \
  } while (0)

  for (int t = 0; t < 512; ++t) {
    // Per-step opaque offset: keeps all weight tile loads inside the loop.
    const size_t z = opaque_zero();

    // ---- stage x_t tile: 16 rows x 64 f32 -> bf16 (4 elems / thread) ----
    {
      int m = tid >> 4;             // 0..15
      int d = (tid & 15) << 2;      // 0..60
      const float4 xv = *(const float4*)(x + ((size_t)(batch0 + m) * 512 + t) * 64 + d);
      unsigned short* dst = &xbuf[m * 64 + d];
      dst[0] = f2bf(xv.x); dst[1] = f2bf(xv.y); dst[2] = f2bf(xv.z); dst[3] = f2bf(xv.w);
    }
    __syncthreads();

    // ================= layer 0: 6 pipelined k-groups =================
    v8f acc[4];
#pragma unroll
    for (int q = 0; q < 4; ++q) {
#pragma unroll
      for (int e = 0; e < 8; ++e) acc[q][e] = bias0[q];
    }
    PIPE_STEP(Wih0,  64, 32, xbuf,   64,  0);
    PIPE_STEP(Whh0, 128,  0, xbuf,   64, 32);
    PIPE_STEP(Whh0, 128, 32, h0buf, 128,  0);
    PIPE_STEP(Whh0, 128, 64, h0buf, 128, 32);
    PIPE_STEP(Whh0, 128, 96, h0buf, 128, 64);
    PIPE_STEP(Wih1, 128,  0, h0buf, 128, 96);   // prefetch layer-1 g0

    // cell math + barriers overlap the in-flight W_ih1 loads
    v8f h0v;
    cell_update(acc, c0, h0v);
    __syncthreads();   // all waves done reading h0buf
#pragma unroll
    for (int r = 0; r < 8; ++r) {
      int m = r + ((lane >> 4) << 3);
      h0buf[m * 128 + ncol] = f2bf(h0v[r]);
    }
    __syncthreads();   // h0buf ready (layer-1 input)

    // ================= layer 1: 8 pipelined k-groups =================
#pragma unroll
    for (int q = 0; q < 4; ++q) {
#pragma unroll
      for (int e = 0; e < 8; ++e) acc[q][e] = bias1[q];
    }
    PIPE_STEP(Wih1, 128, 32, h0buf, 128,  0);
    PIPE_STEP(Wih1, 128, 64, h0buf, 128, 32);
    PIPE_STEP(Wih1, 128, 96, h0buf, 128, 64);
    PIPE_STEP(Whh1, 128,  0, h0buf, 128, 96);
    PIPE_STEP(Whh1, 128, 32, h1buf, 128,  0);
    PIPE_STEP(Whh1, 128, 64, h1buf, 128, 32);
    PIPE_STEP(Whh1, 128, 96, h1buf, 128, 64);
    PIPE_STEP(Wih0,  64,  0, h1buf, 128, 96);   // prefetch next-t layer-0 g0

    // cell math + barriers + x staging overlap the in-flight W_ih0 loads
    v8f h1v;
    cell_update(acc, c1, h1v);
    __syncthreads();   // all waves done reading h1buf
#pragma unroll
    for (int r = 0; r < 8; ++r) {
      int m = r + ((lane >> 4) << 3);
      h1buf[m * 128 + ncol] = f2bf(h1v[r]);
    }
    __syncthreads();   // h1buf ready for next step
  }
#undef PIPE_STEP

  // ---- final linear: out[b] = h2_last[b,:] . W_lin + b_lin ----
  if (tid < 16) {
    float s = 0.f;
#pragma unroll 4
    for (int k = 0; k < 128; ++k) s += bf2f(h1buf[tid * 128 + k]) * Wlin[k];
    out[batch0 + tid] = s + blin[0];
  }
}

extern "C" void kernel_launch(void* const* d_in, const int* in_sizes, int n_in,
                              void* d_out, int out_size, void* d_ws, size_t ws_size,
                              hipStream_t stream) {
  const float* x     = (const float*)d_in[0];
  const float* Wih0  = (const float*)d_in[1];
  const float* Whh0  = (const float*)d_in[2];
  const float* bih0  = (const float*)d_in[3];
  const float* bhh0  = (const float*)d_in[4];
  const float* Wih1  = (const float*)d_in[5];
  const float* Whh1  = (const float*)d_in[6];
  const float* bih1  = (const float*)d_in[7];
  const float* bhh1  = (const float*)d_in[8];
  const float* Wlin  = (const float*)d_in[9];
  const float* blin  = (const float*)d_in[10];

  unsigned short* wbf = (unsigned short*)d_ws;
  unsigned short* Wih0b = wbf;            // 512*64   = 32768
  unsigned short* Whh0b = wbf + 32768;    // 512*128  = 65536
  unsigned short* Wih1b = wbf + 98304;    // 512*128
  unsigned short* Whh1b = wbf + 163840;   // 512*128 -> total 229376 elems

  // one-time bf16 weight conversion (229376 elems / 256 = 896 blocks)
  cvt_weights<<<896, 256, 0, stream>>>(Wih0, Whh0, Wih1, Whh1, wbf);

  // fused 2-layer scan: 32 blocks x 256 threads (8 wave32s), 16 batch rows each
  lstm_scan<<<32, 256, 0, stream>>>(x, Wih0b, Whh0b, Wih1b, Whh1b,
                                    bih0, bhh0, bih1, bhh1,
                                    Wlin, blin, (float*)d_out);
}
